// FlexEncoderLayer_54915451846954
// MI455X (gfx1250) — matrix operand order repaired and checked by reading, not compiled, not run on hardware
//
#include <hip/hip_runtime.h>
#include <cstdint>
#include <cstddef>

// ---------------------------------------------------------------------------
// FlexEncoderLayer for MI455X (gfx1250): bf16 WMMA pipeline, fp32 accumulate,
// Tensor Data Mover (TDM) async LDS staging with double buffering.
// B=4, S=1024, D=1024, H=16, dh=64, DF=2048, M = B*S = 4096.
// ---------------------------------------------------------------------------

#define DEV __device__ __forceinline__

typedef __attribute__((ext_vector_type(16))) __bf16       v16bf;
typedef __attribute__((ext_vector_type(8)))  float        v8f;
typedef __attribute__((ext_vector_type(8)))  unsigned int v8u;
typedef __attribute__((ext_vector_type(4)))  unsigned int v4u;
typedef __attribute__((ext_vector_type(8)))  int          v8i;
typedef __attribute__((ext_vector_type(4)))  int          v4i;

union FragBF { v8u u; v16bf h; };

DEV unsigned short f2bf(float f) {            // fp32 -> bf16, round-nearest-even
    unsigned int u = __float_as_uint(f);
    u += 0x7FFFu + ((u >> 16) & 1u);
    return (unsigned short)(u >> 16);
}

DEV v8f wmma_bf16(const v8u& a, const v8u& b, v8f c) {
    FragBF fa, fb; fa.u = a; fb.u = b;
    return __builtin_amdgcn_wmma_f32_16x16x32_bf16(
        /*neg_a=*/false, fa.h, /*neg_b=*/false, fb.h,
        /*c_mod=*/(short)0, c, /*reuse_a=*/false, /*reuse_b=*/false);
}

// ---------------------------------------------------------------------------
// TDM: async 2D bf16 tile load, global -> LDS. Tile = tile1 rows x tile0
// contiguous elements, row stride = stride0 elements. Descriptor framed so
// tensor == tile (never OOB). D# layout per CDNA5 ISA.
// Issue from ONE wave only; completion tracked by that wave's TENSORcnt.
// ---------------------------------------------------------------------------
DEV void tdm_load_2d_bf16(const void* gtile, unsigned lds_off,
                          unsigned tile0, unsigned tile1, unsigned stride0) {
    const unsigned long long ga = (unsigned long long)(uintptr_t)gtile;
    v4u g0;
    g0[0] = 1u;                                            // count=1 (valid D#)
    g0[1] = lds_off;                                       // lds_addr (bytes)
    g0[2] = (unsigned)(ga & 0xFFFFFFFFull);                // global_addr[31:0]
    g0[3] = (unsigned)((ga >> 32) & 0x1FFFFFFull) |        // global_addr[56:32]
            (2u << 30);                                    // type = 2 ("image")
    v8i g1;
    g1[0] = (int)(1u << 16);                               // data_size=1 -> 2B
    g1[1] = (int)((tile0 & 0xFFFFu) << 16);                // tensor_dim0[15:0]
    g1[2] = (int)((tile0 >> 16) | ((tile1 & 0xFFFFu) << 16)); // dim0 hi|dim1 lo
    g1[3] = (int)((tile1 >> 16) | (tile0 << 16));          // dim1 hi | tile_dim0
    g1[4] = (int)(tile1 & 0xFFFFu);                        // tile_dim1 (tile_dim2=0)
    g1[5] = (int)stride0;                                  // tensor_dim0_stride lo
    g1[6] = 0;                                             // stride0 hi | stride1 lo
    g1[7] = 0;
    v4i g2 = {0, 0, 0, 0};
    v4i g3 = {0, 0, 0, 0};
#if __clang_major__ >= 23
    v8i g4 = {0, 0, 0, 0, 0, 0, 0, 0};
    __builtin_amdgcn_tensor_load_to_lds(g0, g1, g2, g3, g4, 0);
#else
    __builtin_amdgcn_tensor_load_to_lds(g0, g1, g2, g3, 0);
#endif
}

// ---------------------------------------------------------------------------
// fp32 -> bf16 elementwise convert (weights)
// ---------------------------------------------------------------------------
__global__ void cvt_f32_bf16_k(const float* __restrict__ in,
                               unsigned short* __restrict__ out, int n) {
    int i = blockIdx.x * 256 + threadIdx.x;
    if (i < n) out[i] = f2bf(in[i]);
}

// ---------------------------------------------------------------------------
// LayerNorm over D=1024, one block (256 threads) per row, bf16 output
// ---------------------------------------------------------------------------
__global__ __launch_bounds__(256)
void layernorm_k(const float* __restrict__ x, const float* __restrict__ g,
                 const float* __restrict__ b, unsigned short* __restrict__ out) {
    const int row = blockIdx.x;
    const float* xr = x + (size_t)row * 1024;
    __shared__ float red[256];
    float v[4];
    float s = 0.f;
#pragma unroll
    for (int i = 0; i < 4; i++) { v[i] = xr[threadIdx.x + i * 256]; s += v[i]; }
    red[threadIdx.x] = s; __syncthreads();
    for (int off = 128; off > 0; off >>= 1) {
        if (threadIdx.x < off) red[threadIdx.x] += red[threadIdx.x + off];
        __syncthreads();
    }
    const float mu = red[0] * (1.f / 1024.f);
    __syncthreads();
    float s2 = 0.f;
#pragma unroll
    for (int i = 0; i < 4; i++) { float d = v[i] - mu; s2 += d * d; }
    red[threadIdx.x] = s2; __syncthreads();
    for (int off = 128; off > 0; off >>= 1) {
        if (threadIdx.x < off) red[threadIdx.x] += red[threadIdx.x + off];
        __syncthreads();
    }
    const float rstd = rsqrtf(red[0] * (1.f / 1024.f) + 1e-5f);
#pragma unroll
    for (int i = 0; i < 4; i++) {
        const int idx = threadIdx.x + i * 256;
        out[(size_t)row * 1024 + idx] = f2bf((v[i] - mu) * rstd * g[idx] + b[idx]);
    }
}

// ---------------------------------------------------------------------------
// Tiled bf16 WMMA GEMM:  C[M x N] = A[M x K] * W[N x K]^T + bias
// Block tile 128x64, 256 threads = 8 waves; wave w owns rows [w*16, w*16+16)
// of the block tile -> 4 accumulator tiles of 16x16 (v8f each).
// K-panels staged via TDM (double buffered): wave 0 issues tensor_load_to_lds
// for panel i+1 while all waves run WMMAs on panel i; gate = s_wait_tensorcnt.
// Epilogues: EP=0 QKV split->bf16 [B,H,S,dh]; EP=1 +residual -> fp32;
//            EP=2 exact GELU -> bf16.
// LDS map (base assumed 0): A0 @0 (8KB), A1 @8192, B0 @16384 (4KB), B1 @20480.
// ---------------------------------------------------------------------------
template <int EP>
__global__ __launch_bounds__(256)
void gemm_bf16_k(const unsigned short* __restrict__ A,
                 const unsigned short* __restrict__ W,   // [N][K] bf16
                 const float* __restrict__ bias,
                 int K, int N,
                 unsigned short* __restrict__ o0,
                 unsigned short* __restrict__ o1,
                 unsigned short* __restrict__ o2,
                 float* __restrict__ outf,
                 const float* __restrict__ res) {
    __shared__ unsigned int smem[2 * 2048 + 2 * 1024];   // A0,A1,B0,B1
    const int tid  = threadIdx.x;
    const int wave = tid >> 5, lane = tid & 31;
    const int l15  = lane & 15;
    const int hk   = (lane & 16) ? 4 : 0;   // A-frag K-octet select per lane half
    const int hb   = (lane & 16) ? 8 : 0;   // unused: B-frag also indexes with hk
    const int m0 = blockIdx.y * 128, n0 = blockIdx.x * 64;
    const unsigned short* Ab = A + (size_t)m0 * K;       // tile row origin
    const unsigned short* Wb = W + (size_t)n0 * K;
    v8f acc[4] = {};

    const int nsteps = K >> 5;
    if (tid < 32) {                                       // prologue: panel 0
        tdm_load_2d_bf16(Ab, 0u,     32u, 128u, (unsigned)K);
        tdm_load_2d_bf16(Wb, 16384u, 32u,  64u, (unsigned)K);
    }
    for (int i = 0; i < nsteps; i++) {
        const int cur = i & 1;
        if (tid < 32) {
            if (i + 1 < nsteps) {                          // issue panel i+1
                const int k1 = (i + 1) << 5;
                tdm_load_2d_bf16(Ab + k1, cur ? 0u : 8192u,      32u, 128u, (unsigned)K);
                tdm_load_2d_bf16(Wb + k1, cur ? 16384u : 20480u, 32u,  64u, (unsigned)K);
                __builtin_amdgcn_s_wait_tensorcnt(2);      // panel i landed
            } else {
                __builtin_amdgcn_s_wait_tensorcnt(0);
            }
        }
        __syncthreads();                                   // publish panel i
        const unsigned int* lAc = smem + (cur ? 2048 : 0);
        const unsigned int* lBc = smem + 4096 + (cur ? 1024 : 0);
        v8u a;
        const unsigned int* la = lAc + (wave * 16 + l15) * 16;
#pragma unroll
        for (int j = 0; j < 4; j++) { a[j] = la[hk + j]; a[4 + j] = la[8 + hk + j]; }
#pragma unroll
        for (int nt = 0; nt < 4; nt++) {
            v8u b;
            const unsigned int* lb = lBc + (nt * 16 + l15) * 16;
#pragma unroll
            for (int j = 0; j < 8; j++) b[j] = lb[hk + (j & 3) + ((j >> 2) << 3)];
            acc[nt] = wmma_bf16(a, b, acc[nt]);
        }
        __syncthreads();            // all waves done with panel i -> reusable
    }

    const int rbase = m0 + wave * 16 + ((lane & 16) ? 8 : 0);
#pragma unroll
    for (int nt = 0; nt < 4; nt++) {
        const int n = n0 + nt * 16 + l15;
        const float bv = bias[n];
#pragma unroll
        for (int r = 0; r < 8; r++) {
            float vv = acc[nt][r] + bv;
            const int row = rbase + r;
            if constexpr (EP == 0) {
                // scatter qkv: col n -> (part, head, dd); row -> (b, s)
                const int part = n >> 10, d = n & 1023, head = d >> 6, dd = d & 63;
                const int bb = row >> 10, ss = row & 1023;
                const size_t di = (((size_t)(bb * 16 + head)) * 1024 + ss) * 64 + dd;
                const unsigned short val = f2bf(vv);
                if (part == 0) o0[di] = val;
                else if (part == 1) o1[di] = val;
                else o2[di] = val;
            } else if constexpr (EP == 1) {
                const size_t di = (size_t)row * N + n;
                outf[di] = res[di] + vv;
            } else {
                const float ge = 0.5f * vv * (1.0f + erff(vv * 0.70710678118f));
                o0[(size_t)row * N + n] = f2bf(ge);
            }
        }
    }
}

// ---------------------------------------------------------------------------
// Flash attention with logit soft-cap + causal mask.
// q,k,v: bf16 [B*H][S][64]. o: bf16 [B][S][D] (heads re-interleaved).
// Block: 128 thr = 4 waves; each block does 64 q-rows of one (b,h);
// each wave owns a 16-row band. K tile arrives via TDM (overlapped with the
// manual transposed V staging); online softmax row reductions use
// __shfl_xor(width=16) since a C-tile row lives in one 16-lane half.
// LDS map (base 0): lK @0 (8KB), lVT @8192 (8KB), lP @16384 (8KB).
// ---------------------------------------------------------------------------
__global__ __launch_bounds__(128)
void attn_k(const unsigned short* __restrict__ q,
            const unsigned short* __restrict__ k,
            const unsigned short* __restrict__ v,
            unsigned short* __restrict__ o) {
    __shared__ unsigned int smem[3 * 2048];
    unsigned int*   lK  = smem;                              // [key][d-pair]
    unsigned short* lVT = (unsigned short*)(smem + 2048);    // [d][key]
    unsigned short* lP  = (unsigned short*)(smem + 4096);    // per-wave P
    const int tid  = threadIdx.x;
    const int wave = tid >> 5, lane = tid & 31;
    const int l15  = lane & 15;
    const int hk   = (lane & 16) ? 4 : 0;
    const int hb   = (lane & 16) ? 8 : 0;
    const int bh = blockIdx.y, qt = blockIdx.x;
    const size_t base32 = (size_t)bh * 1024 * 32;   // u32 per (b,h) = S*64/2
    const int q0 = qt * 64;

    // Q fragments for this wave's 16-row band: 2 K-steps of 32 over dh=64
    v8u qa[2];
    {
        const unsigned int* Qg = (const unsigned int*)q + base32 +
                                 (size_t)(q0 + wave * 16 + l15) * 32;
#pragma unroll
        for (int ks = 0; ks < 2; ks++)
#pragma unroll
            for (int j = 0; j < 4; j++) {
                qa[ks][j]     = Qg[ks * 16 + hk + j];
                qa[ks][4 + j] = Qg[ks * 16 + 8 + hk + j];
            }
    }

    float mI[8], lI[8];
#pragma unroll
    for (int r = 0; r < 8; r++) { mI[r] = -1e30f; lI[r] = 0.f; }
    v8f oacc[4] = {};

    for (int kt = 0; kt <= qt; kt++) {      // causal: only lower-triangular tiles
        // K tile via TDM (64 rows x 64 bf16 contiguous)
        if (tid < 32)
            tdm_load_2d_bf16((const unsigned short*)k + base32 * 2 +
                                 (size_t)kt * 64 * 64,
                             0u, 64u, 64u, 64u);
        // V tile staged transposed into [d][key] by all threads (overlaps TDM)
        const unsigned int* Vg = (const unsigned int*)v + base32 + (size_t)kt * 64 * 32;
#pragma unroll
        for (int i = 0; i < 16; i++) {
            const int idx = tid + i * 128;
            const unsigned int u = Vg[idx];
            const int key = idx >> 5, dp = idx & 31;
            lVT[(2 * dp) * 64 + key]     = (unsigned short)(u & 0xFFFFu);
            lVT[(2 * dp + 1) * 64 + key] = (unsigned short)(u >> 16);
        }
        if (tid < 32) __builtin_amdgcn_s_wait_tensorcnt(0);
        __syncthreads();

        // scores = Q * K^T  (B-frag reads K rows: B[dk][n] = K[n][dk])
        v8f sc[4] = {};
#pragma unroll
        for (int ks = 0; ks < 2; ks++) {
#pragma unroll
            for (int nt = 0; nt < 4; nt++) {
                v8u b;
                const unsigned int* lb = lK + (nt * 16 + l15) * 32 + ks * 16;
#pragma unroll
                for (int j = 0; j < 8; j++) b[j] = lb[hk + (j & 3) + ((j >> 2) << 3)];
                sc[nt] = wmma_bf16(qa[ks], b, sc[nt]);
            }
        }

        // scale 1/sqrt(64), soft-cap 6*tanh(s/6), causal mask
        const int rowg0 = q0 + wave * 16 + ((lane & 16) ? 8 : 0);
#pragma unroll
        for (int nt = 0; nt < 4; nt++) {
            const int col = kt * 64 + nt * 16 + l15;
#pragma unroll
            for (int r = 0; r < 8; r++) {
                float s = sc[nt][r] * 0.125f;
                s = 6.0f * tanhf(s * (1.0f / 6.0f));
                if (col > rowg0 + r) s = -1e9f;
                sc[nt][r] = s;
            }
        }

        // online softmax update
        float rsum[8], alpha[8];
#pragma unroll
        for (int r = 0; r < 8; r++) {
            float m = fmaxf(fmaxf(sc[0][r], sc[1][r]), fmaxf(sc[2][r], sc[3][r]));
#pragma unroll
            for (int d = 1; d < 16; d <<= 1) m = fmaxf(m, __shfl_xor(m, d, 16));
            const float mn = fmaxf(mI[r], m);
            alpha[r] = __expf(mI[r] - mn);
            mI[r] = mn;
            rsum[r] = 0.f;
        }
#pragma unroll
        for (int nt = 0; nt < 4; nt++)
#pragma unroll
            for (int r = 0; r < 8; r++) {
                const float p = __expf(sc[nt][r] - mI[r]);
                sc[nt][r] = p;
                rsum[r] += p;
            }
#pragma unroll
        for (int r = 0; r < 8; r++) {
            float s = rsum[r];
#pragma unroll
            for (int d = 1; d < 16; d <<= 1) s += __shfl_xor(s, d, 16);
            lI[r] = lI[r] * alpha[r] + s;
#pragma unroll
            for (int nt = 0; nt < 4; nt++) oacc[nt][r] *= alpha[r];
        }

        // stage P (C-layout regs -> row-major LDS) to re-read as A-fragments
        unsigned short* lp = lP + wave * (16 * 64);
        const int prow = (lane & 16) ? 8 : 0;
#pragma unroll
        for (int nt = 0; nt < 4; nt++)
#pragma unroll
            for (int r = 0; r < 8; r++)
                lp[(r + prow) * 64 + nt * 16 + l15] = f2bf(sc[nt][r]);
        __syncthreads();

        // O += P * V  (contraction over 64 keys, 2 K-steps of 32)
        const unsigned int* lpu = (const unsigned int*)lp + l15 * 32;
#pragma unroll
        for (int ks = 0; ks < 2; ks++) {
            v8u pa;
#pragma unroll
            for (int j = 0; j < 4; j++) {
                pa[j]     = lpu[ks * 16 + hk + j];
                pa[4 + j] = lpu[ks * 16 + 8 + hk + j];
            }
#pragma unroll
            for (int nt = 0; nt < 4; nt++) {
                v8u b;
                const unsigned int* lb = (const unsigned int*)lVT +
                                         (nt * 16 + l15) * 32 + ks * 16;
#pragma unroll
                for (int j = 0; j < 8; j++) b[j] = lb[hk + (j & 3) + ((j >> 2) << 3)];
                oacc[nt] = wmma_bf16(pa, b, oacc[nt]);
            }
        }
        __syncthreads();   // protect lK/lVT/lP before next iteration
    }

    // normalize and write: o[b][s][h*64 + d]  (== transpose(0,2,1,3).reshape)
    const int b_ = bh >> 4, h_ = bh & 15;
    const int rowl0 = wave * 16 + ((lane & 16) ? 8 : 0);
#pragma unroll
    for (int r = 0; r < 8; r++) {
        const float inv = 1.0f / lI[r];
        const int srow = q0 + rowl0 + r;
#pragma unroll
        for (int nt = 0; nt < 4; nt++)
            o[((size_t)(b_ * 1024 + srow)) * 1024 + h_ * 64 + nt * 16 + l15] =
                f2bf(oacc[nt][r] * inv);
    }
}

// ---------------------------------------------------------------------------
// launch
// ---------------------------------------------------------------------------
extern "C" void kernel_launch(void* const* d_in, const int* in_sizes, int n_in,
                              void* d_out, int out_size, void* d_ws, size_t ws_size,
                              hipStream_t stream) {
    (void)in_sizes; (void)n_in; (void)out_size; (void)ws_size;
    const float* x     = (const float*)d_in[0];
    /* d_in[1] = bool mask, unused: causal mask computed analytically */
    const float* qkv_w = (const float*)d_in[2];
    const float* qkv_b = (const float*)d_in[3];
    const float* out_w = (const float*)d_in[4];
    const float* out_b = (const float*)d_in[5];
    const float* ln1_g = (const float*)d_in[6];
    const float* ln1_b = (const float*)d_in[7];
    const float* ln2_g = (const float*)d_in[8];
    const float* ln2_b = (const float*)d_in[9];
    const float* ff1_w = (const float*)d_in[10];
    const float* ff1_b = (const float*)d_in[11];
    const float* ff2_w = (const float*)d_in[12];
    const float* ff2_b = (const float*)d_in[13];
    float* out = (float*)d_out;

    const int M = 4096, D = 1024, DF = 2048, N3 = 3072;

    char* ws = (char*)d_ws;
    size_t off = 0;
    auto alloc = [&](size_t bytes) -> void* {
        void* p = ws + off; off += (bytes + 255) & ~(size_t)255; return p;
    };
    unsigned short* hbf   = (unsigned short*)alloc((size_t)M  * D  * 2);
    unsigned short* qkvwb = (unsigned short*)alloc((size_t)N3 * D  * 2);
    unsigned short* outwb = (unsigned short*)alloc((size_t)D  * D  * 2);
    unsigned short* ff1wb = (unsigned short*)alloc((size_t)DF * D  * 2);
    unsigned short* ff2wb = (unsigned short*)alloc((size_t)D  * DF * 2);
    unsigned short* qb    = (unsigned short*)alloc((size_t)M  * D  * 2);
    unsigned short* kb    = (unsigned short*)alloc((size_t)M  * D  * 2);
    unsigned short* vb    = (unsigned short*)alloc((size_t)M  * D  * 2);
    unsigned short* ob    = (unsigned short*)alloc((size_t)M  * D  * 2);
    float*          x2    = (float*)         alloc((size_t)M  * D  * 4);
    unsigned short* h2b   = (unsigned short*)alloc((size_t)M  * D  * 2);
    unsigned short* ffb   = (unsigned short*)alloc((size_t)M  * DF * 2);

    // weight fp32 -> bf16
    cvt_f32_bf16_k<<<(N3 * D + 255) / 256, 256, 0, stream>>>(qkv_w, qkvwb, N3 * D);
    cvt_f32_bf16_k<<<(D  * D + 255) / 256, 256, 0, stream>>>(out_w, outwb, D * D);
    cvt_f32_bf16_k<<<(DF * D + 255) / 256, 256, 0, stream>>>(ff1_w, ff1wb, DF * D);
    cvt_f32_bf16_k<<<(D * DF + 255) / 256, 256, 0, stream>>>(ff2_w, ff2wb, D * DF);

    // LN1
    layernorm_k<<<M, 256, 0, stream>>>(x, ln1_g, ln1_b, hbf);
    // QKV projection, scattered to [B,H,S,dh] bf16
    gemm_bf16_k<0><<<dim3(N3 / 64, M / 128), 256, 0, stream>>>(
        hbf, qkvwb, qkv_b, D, N3, qb, kb, vb, nullptr, nullptr);
    // attention
    attn_k<<<dim3(1024 / 64, 4 * 16), 128, 0, stream>>>(qb, kb, vb, ob);
    // out projection + residual(x) -> x2 (fp32)
    gemm_bf16_k<1><<<dim3(D / 64, M / 128), 256, 0, stream>>>(
        ob, outwb, out_b, D, D, nullptr, nullptr, nullptr, x2, x);
    // LN2
    layernorm_k<<<M, 256, 0, stream>>>(x2, ln2_g, ln2_b, h2b);
    // FF1 + exact GELU -> bf16
    gemm_bf16_k<2><<<dim3(DF / 64, M / 128), 256, 0, stream>>>(
        h2b, ff1wb, ff1_b, D, DF, ffb, nullptr, nullptr, nullptr, nullptr);
    // FF2 + residual(x2) -> out (fp32)
    gemm_bf16_k<1><<<dim3(D / 64, M / 128), 256, 0, stream>>>(
        ffb, ff2wb, ff2_b, DF, D, nullptr, nullptr, nullptr, out, x2);
}
